// Efficient_TemporalGRU_21715354648830
// MI455X (gfx1250) — compile-verified, matrix operand order checked by reading, not run on hardware
//
#include <hip/hip_runtime.h>
#include <hip/hip_bf16.h>

// ---------------------------------------------------------------------------
// Sizes fixed by the reference
// ---------------------------------------------------------------------------
#define BB   4
#define TT   10
#define CC   32
#define HW   65536          // 256*256
#define RHH  60
// GRU GEMM padded: din 92 -> 96 (3 K-slabs of 32), RH 60 -> 64 (4 out tiles)
// cin slot 95 (hidden pad k=63) carries a constant 1.0 -> biases live in weights
// projection: 2*RH 120 -> 128 (4 K-slabs), C=32 (2 out tiles); bp baked at K=63

typedef __attribute__((ext_vector_type(16))) __bf16        v16bf;
typedef __attribute__((ext_vector_type(8)))  float         v8f;
typedef __attribute__((ext_vector_type(8)))  unsigned int  v8u;

// K placement for 16-bit 16x16x32 WMMA operands: lane half hh, vgpr slot v
__device__ __forceinline__ int kmap(int v, int h) {
    return (v < 4) ? (8 * h + 2 * v) : (16 + 8 * h + 2 * (v - 4));
}
__device__ __forceinline__ unsigned int pack2bf16(float a, float b) {
    unsigned int ua = __float_as_uint(a) + 0x8000u;
    unsigned int ub = __float_as_uint(b) + 0x8000u;
    return __builtin_amdgcn_perm(ub, ua, 0x07060302u);   // {ub[31:16], ua[31:16]}
}
__device__ __forceinline__ unsigned short bf16r(float f) {
    return (unsigned short)((__float_as_uint(f) + 0x8000u) >> 16);
}
__device__ __forceinline__ v16bf as_bf(v8u u) {
    union { v8u u; v16bf b; } c; c.u = u; return c.b;
}
__device__ __forceinline__ v8f splat8(float f) {
    v8f r;
#pragma unroll
    for (int i = 0; i < 8; i++) r[i] = f;
    return r;
}
__device__ __forceinline__ float fsigmoid(float x) {
    return __builtin_amdgcn_rcpf(1.f + __expf(-x));      // v_exp_f32 + v_rcp_f32
}
__device__ __forceinline__ v8f wmma_bf16(v16bf a, v16bf b, v8f c) {
    return __builtin_amdgcn_wmma_f32_16x16x32_bf16(false, a, false, b,
                                                   (short)0, c, false, false);
}

// ---------------------------------------------------------------------------
// Weight prep: padded bf16 B-tiles in lane layout, biases baked into the
// constant-one row. ws: 36 gate tiles (g,s,nt) + 8 proj tiles (s,nt), 1KB each.
// ---------------------------------------------------------------------------
__global__ __launch_bounds__(32) void prep_weights(
    const float* __restrict__ Wu, const float* __restrict__ Wr,
    const float* __restrict__ Wc, const float* __restrict__ Wp,
    const float* __restrict__ bu, const float* __restrict__ br,
    const float* __restrict__ bc, const float* __restrict__ bp,
    unsigned int* __restrict__ wt)
{
    const int tileId = blockIdx.x;            // 0..43
    const int lane   = threadIdx.x;           // 0..31
    const int hh = lane >> 4, n = lane & 15;  // B column
    unsigned int* dst = wt + tileId * 256 + lane * 8;

    if (tileId < 36) {
        const int g = tileId / 12, s = (tileId / 4) % 3, nt = tileId % 4;
        const float* W = (g == 0) ? Wu : ((g == 1) ? Wr : Wc);
        const float* Bv = (g == 0) ? bu : ((g == 1) ? br : bc);
        const int out = nt * 16 + n;
#pragma unroll
        for (int v = 0; v < 8; v++) {
            const int k = kmap(v, hh);        // even, 0..30 within slab
            float f0 = 0.f, f1 = 0.f;
            if (out < RHH) {
                if (s == 0) {                 // x channels (all valid, C=32)
                    f0 = W[out * 92 + k];
                    f1 = W[out * 92 + k + 1];
                } else {                      // hidden part; j = hidden index
                    const int j0 = (s - 1) * 32 + k, j1 = j0 + 1;
                    if (j0 < RHH) f0 = W[out * 92 + 32 + j0];
                    if (j1 < RHH) f1 = W[out * 92 + 32 + j1];
                    else if (j1 == 63) f1 = Bv[out];     // bias row (A[k63]==1)
                }
            }
            dst[v] = pack2bf16(f0, f1);
        }
    } else {
        const int pt = tileId - 36, s = pt / 2, nt = pt % 2;
        const int out = nt * 16 + n;          // output channel < 32
#pragma unroll
        for (int v = 0; v < 8; v++) {
            const int kg = 32 * s + kmap(v, hh);
#pragma unroll
            for (int q = 0; q < 2; q++) {
                const int k = kg + q;
                float f = 0.f;
                if (k < RHH)                f = Wp[out * 120 + k];          // h_fwd
                else if (k == 63)           f = bp[out];                    // bias row
                else if (k >= 64 && k < 64 + RHH) f = Wp[out * 120 + 60 + (k - 64)]; // h_bwd
                if (q == 0) dst[v]  = (unsigned int)bf16r(f);
                else        dst[v] |= ((unsigned int)bf16r(f)) << 16;
            }
        }
    }
}

// ---------------------------------------------------------------------------
// Main kernel: one wave = 16 pixel sequences; 4 waves/block.
// LDS: 36KB gate weights (block-shared) + 6KB/wave h staging.
// ---------------------------------------------------------------------------
__device__ __forceinline__ v16bf loadA_h(const unsigned short* buf, int sb, int lane) {
    const int hh = lane >> 4, m = lane & 15;
    const unsigned char* p = (const unsigned char*)buf + m * 128 + sb * 64 + hh * 16;
    const uint4 lo = *(const uint4*)(p);
    const uint4 hi = *(const uint4*)(p + 32);
    v8u u; u[0]=lo.x; u[1]=lo.y; u[2]=lo.z; u[3]=lo.w;
           u[4]=hi.x; u[5]=hi.y; u[6]=hi.z; u[7]=hi.w;
    return as_bf(u);
}
__device__ __forceinline__ void store_h(unsigned short* buf, const v8f* h, int lane) {
    const int hh = lane >> 4, n = lane & 15;
#pragma unroll
    for (int nt = 0; nt < 4; nt++)
#pragma unroll
        for (int v = 0; v < 8; v++)
            buf[(v + 8 * hh) * 64 + nt * 16 + n] = bf16r(h[nt][v]);
}
__device__ __forceinline__ v16bf ldWlds(const unsigned char* base, int idx,
                                        int lane, unsigned off) {
    return as_bf(*(const v8u*)(base + idx * 1024 + lane * 32 + off));
}
__device__ __forceinline__ v16bf ldWg(const unsigned int* wt, int idx, int lane) {
    return as_bf(*(const v8u*)(wt + idx * 256 + lane * 8));
}
__device__ __forceinline__ void loadX(float* xf, const float* x, int b, int tr,
                                      int pm, int hh) {
    const float* xb = x + ((size_t)(b * TT + tr) * CC) * HW + pm;
#pragma unroll
    for (int v = 0; v < 8; v++) {
        const int c = kmap(v, hh);
        xf[2 * v]     = xb[(size_t)c * HW];
        xf[2 * v + 1] = xb[(size_t)(c + 1) * HW];
    }
}

__global__ __launch_bounds__(128, 4) void gru_main(
    const float* __restrict__ x,
    const unsigned int* __restrict__ wt,
    float* __restrict__ out)
{
    __shared__ __align__(16) unsigned char sW[36864];     // 36 gate tiles
    __shared__ __align__(16) unsigned char sStage[4 * 6144];

    const int wave = threadIdx.x >> 5;
    const int lane = threadIdx.x & 31;
    const int tile = blockIdx.x * 4 + wave;
    const int hh = lane >> 4, m = lane & 15;
    const int b  = tile >> 12;                 // tile / (HW/16)
    const int p0 = (tile & 4095) << 4;
    const bool lastlane = (m == 15);           // holds padded out column 63

    // cooperative copy of gate weight tiles to LDS (once per block)
    {
        const uint4* src = (const uint4*)wt;
        uint4* dst = (uint4*)sW;
        for (int i = threadIdx.x; i < 36864 / 16; i += 128) dst[i] = src[i];
    }
    __syncthreads();

    unsigned char* myLds = sStage + wave * 6144;
    unsigned short* bufF = (unsigned short*)(myLds);
    unsigned short* bufR = (unsigned short*)(myLds + 2048);
    unsigned short* bufB = (unsigned short*)(myLds + 4096);

    const v8f zero = splat8(0.f);

#pragma unroll 1
    for (int d = 0; d < 2; d++) {
        unsigned short* bufH = d ? bufB : bufF;
        v8f h[4];
#pragma unroll
        for (int nt = 0; nt < 4; nt++) h[nt] = zero;
#pragma unroll
        for (int v = 0; v < 8; v++) h[3][v] = lastlane ? 1.f : 0.f;  // k63 = 1
        store_h(bufH, h, lane);
        __threadfence_block();

        float xf[16];
        loadX(xf, x, b, d ? TT - 1 : 0, p0 + m, hh);

#pragma unroll 1
        for (int t = 0; t < TT; t++) {
            unsigned woff = 0;
            asm volatile("" : "+v"(woff));     // defeat LICM on LDS weight loads

            // pack current x A-tile (loads issued last iteration)
            v8u axu;
#pragma unroll
            for (int v = 0; v < 8; v++) axu[v] = pack2bf16(xf[2 * v], xf[2 * v + 1]);
            const v16bf ax  = as_bf(axu);
            const v16bf ah0 = loadA_h(bufH, 0, lane);
            const v16bf ah1 = loadA_h(bufH, 1, lane);

            // update & reset gates: 24 WMMAs (biases ride the k63==1 row)
            v8f ug[4], rg[4];
#pragma unroll
            for (int nt = 0; nt < 4; nt++) {
                ug[nt] = wmma_bf16(ax,  ldWlds(sW, (0*3+0)*4 + nt, lane, woff), zero);
                ug[nt] = wmma_bf16(ah0, ldWlds(sW, (0*3+1)*4 + nt, lane, woff), ug[nt]);
                ug[nt] = wmma_bf16(ah1, ldWlds(sW, (0*3+2)*4 + nt, lane, woff), ug[nt]);
                rg[nt] = wmma_bf16(ax,  ldWlds(sW, (1*3+0)*4 + nt, lane, woff), zero);
                rg[nt] = wmma_bf16(ah0, ldWlds(sW, (1*3+1)*4 + nt, lane, woff), rg[nt]);
                rg[nt] = wmma_bf16(ah1, ldWlds(sW, (1*3+2)*4 + nt, lane, woff), rg[nt]);
            }

            // software pipeline: issue next timestep's x loads now
            if (t + 1 < TT) {
                const int tn = d ? (TT - 2 - t) : (t + 1);
                loadX(xf, x, b, tn, p0 + m, hh);
                __builtin_prefetch(x + ((size_t)(b * TT + tn) * CC) * HW + p0, 0, 0);
            }

#pragma unroll
            for (int nt = 0; nt < 4; nt++)
#pragma unroll
                for (int v = 0; v < 8; v++) {
                    ug[nt][v] = fsigmoid(ug[nt][v]);
                    rg[nt][v] = fsigmoid(rg[nt][v]);
                }

            // h*r, keep k63 pinned to 1.0 so candidate bias row works
            v8f hr[4];
#pragma unroll
            for (int nt = 0; nt < 4; nt++)
#pragma unroll
                for (int v = 0; v < 8; v++) hr[nt][v] = h[nt][v] * rg[nt][v];
#pragma unroll
            for (int v = 0; v < 8; v++) hr[3][v] = lastlane ? 1.f : hr[3][v];
            store_h(bufR, hr, lane);
            __threadfence_block();
            const v16bf ar0 = loadA_h(bufR, 0, lane);
            const v16bf ar1 = loadA_h(bufR, 1, lane);

            // candidate: 12 WMMAs + relu
            v8f cg[4];
#pragma unroll
            for (int nt = 0; nt < 4; nt++) {
                cg[nt] = wmma_bf16(ax,  ldWlds(sW, (2*3+0)*4 + nt, lane, woff), zero);
                cg[nt] = wmma_bf16(ar0, ldWlds(sW, (2*3+1)*4 + nt, lane, woff), cg[nt]);
                cg[nt] = wmma_bf16(ar1, ldWlds(sW, (2*3+2)*4 + nt, lane, woff), cg[nt]);
            }

            // GRU blend; re-pin k63 = 1.0
#pragma unroll
            for (int nt = 0; nt < 4; nt++)
#pragma unroll
                for (int v = 0; v < 8; v++) {
                    const float uu = ug[nt][v];
                    const float cc = fmaxf(cg[nt][v], 0.f);
                    h[nt][v] = (1.f - uu) * h[nt][v] + uu * cc;
                }
#pragma unroll
            for (int v = 0; v < 8; v++) h[3][v] = lastlane ? 1.f : h[3][v];

            store_h(bufH, h, lane);
            __threadfence_block();
        }
    }

    // ---- projection: [h_fwd | h_bwd] (16x128 bf16) @ Wp^T, bp baked at K=63
    const v16bf g0 = loadA_h(bufF, 0, lane);
    const v16bf g1 = loadA_h(bufF, 1, lane);
    const v16bf g2 = loadA_h(bufB, 0, lane);
    const v16bf g3 = loadA_h(bufB, 1, lane);

    v8f o[2];
#pragma unroll
    for (int nt = 0; nt < 2; nt++) {
        o[nt] = wmma_bf16(g0, ldWg(wt, 36 + 0*2 + nt, lane), zero);
        o[nt] = wmma_bf16(g1, ldWg(wt, 36 + 1*2 + nt, lane), o[nt]);
        o[nt] = wmma_bf16(g2, ldWg(wt, 36 + 2*2 + nt, lane), o[nt]);
        o[nt] = wmma_bf16(g3, ldWg(wt, 36 + 3*2 + nt, lane), o[nt]);
    }

    // out[b, ch, p0 + pixel]; ch = (lane&15)+16*nt, pixel = v + 8*hh
#pragma unroll
    for (int nt = 0; nt < 2; nt++) {
        float* ob = out + ((size_t)b * CC + (m + 16 * nt)) * HW + p0 + 8 * hh;
#pragma unroll
        for (int v = 0; v < 8; v++) ob[v] = o[nt][v];
    }
}

// ---------------------------------------------------------------------------
// Host launcher
// ---------------------------------------------------------------------------
extern "C" void kernel_launch(void* const* d_in, const int* in_sizes, int n_in,
                              void* d_out, int out_size, void* d_ws, size_t ws_size,
                              hipStream_t stream) {
    const float* x  = (const float*)d_in[0];
    const float* Wu = (const float*)d_in[1];
    const float* bu = (const float*)d_in[2];
    const float* Wr = (const float*)d_in[3];
    const float* br = (const float*)d_in[4];
    const float* Wc = (const float*)d_in[5];
    const float* bc = (const float*)d_in[6];
    const float* Wp = (const float*)d_in[7];
    const float* bp = (const float*)d_in[8];
    unsigned int* wt = (unsigned int*)d_ws;   // 44 KB packed bf16 weight tiles
    float* out = (float*)d_out;

    hipLaunchKernelGGL(prep_weights, dim3(44), dim3(32), 0, stream,
                       Wu, Wr, Wc, Wp, bu, br, bc, bp, wt);

    // 262144 sequences / 16 per wave = 16384 waves; 4 waves per block
    hipLaunchKernelGGL(gru_main, dim3(4096), dim3(128), 0, stream, x, wt, out);
}
